// attention_9560597200915
// MI455X (gfx1250) — compile-verified, matrix operand order checked by reading, not run on hardware
//
#include <hip/hip_runtime.h>

// ---------------------------------------------------------------------------
// Relative-position attention (Shaw-style), CDNA5 / gfx1250, wave32 WMMA f16.
//   B=4, S=1024, D=512, HEADS=8, HDIM=64, MAX_PEMB=512
// All matmuls via v_wmma_f32_16x16x32_f16. Intermediates f16 (L2-resident).
// pos_attn handled via QP = Q @ pemb^T precompute + clamped gather.
// Round 2: 2 M-tiles/wave in GEMMs, 4 query-tiles/wave in scores,
//          2 query-tiles/wave in P@V -> more WMMA per b128 operand load.
// ---------------------------------------------------------------------------

typedef __attribute__((ext_vector_type(8)))  _Float16 v8h;
typedef __attribute__((ext_vector_type(16))) _Float16 v16h;
typedef __attribute__((ext_vector_type(8)))  float    v8f;

#define BATCH 4
#define SEQ   1024
#define DMODEL 512
#define HEADS 8
#define HDIM  64
#define RNUM  1025   // 2*MAX_PEMB + 1
#define RPAD  1040   // padded to multiple of 16

// ---- WMMA helpers ---------------------------------------------------------

// Load one 16x32 f16 operand tile in A-layout (B operand: pass B^T rows).
// Lane L (row = L&15) loads halves [kbase, kbase+8) and [16+kbase, 16+kbase+8),
// kbase = (L>>4)*8. p points at row base (+kc); caller adds row*stride.
__device__ inline v16h ld_tile(const _Float16* __restrict__ p, int hi) {
    v8h lo = *(const v8h*)(p + hi * 8);
    v8h hh = *(const v8h*)(p + 16 + hi * 8);
    return __builtin_shufflevector(lo, hh, 0,1,2,3,4,5,6,7,8,9,10,11,12,13,14,15);
}

__device__ inline v8f wmma_f16(v16h a, v16h b, v8f c) {
    return __builtin_amdgcn_wmma_f32_16x16x32_f16(
        /*neg_a=*/false, a, /*neg_b=*/false, b,
        /*c_mod=*/(short)0, c, /*reuse_a=*/false, /*reuse_b=*/false);
}

// ---- prep kernels ---------------------------------------------------------

__global__ void cast_f32_to_f16(const float* __restrict__ in,
                                _Float16* __restrict__ out, int n) {
    int i = blockIdx.x * blockDim.x + threadIdx.x;
    if (i < n) out[i] = (_Float16)in[i];
}

// out[n*K + k] = (f16) in[k*N + n]   (W [K,N] -> W^T [N,K])
__global__ void transpose_to_f16(const float* __restrict__ in,
                                 _Float16* __restrict__ out, int K, int N) {
    int i = blockIdx.x * blockDim.x + threadIdx.x;
    if (i >= K * N) return;
    int k = i % K, n = i / K;
    out[i] = (_Float16)in[k * N + n];
}

// pemb [1025,64] f32 -> f16, zero-padded to [1040,64]
__global__ void pemb_prep(const float* __restrict__ pemb,
                          _Float16* __restrict__ out) {
    int i = blockIdx.x * blockDim.x + threadIdx.x;
    if (i >= RPAD * HDIM) return;
    int r = i >> 6;
    out[i] = (r < RNUM) ? (_Float16)pemb[i] : (_Float16)0.0f;
}

// Qh[tok, h*64+d], KVh[tok, {K|V} h*64+d]  ->  per-head layouts:
//   Qr[bh, s, d], Kr[bh, s, d], Vt[bh, d, s]
__global__ void rearrange_qkv(const _Float16* __restrict__ Qh,
                              const _Float16* __restrict__ KVh,
                              _Float16* __restrict__ Qr,
                              _Float16* __restrict__ Kr,
                              _Float16* __restrict__ Vt) {
    int i = blockIdx.x * blockDim.x + threadIdx.x;
    if (i >= BATCH * HEADS * SEQ * HDIM) return;
    int d = i & 63;
    int s = (i >> 6) & (SEQ - 1);
    int h = (i >> 16) & (HEADS - 1);
    int b = i >> 19;
    long tok = (long)b * SEQ + s;
    Qr[i] = Qh[tok * DMODEL + h * HDIM + d];
    Kr[i] = KVh[tok * (2 * DMODEL) + h * HDIM + d];
    Vt[(((long)(b * HEADS + h) * HDIM + d) << 10) + s] =
        KVh[tok * (2 * DMODEL) + DMODEL + h * HDIM + d];
}

// ---- generic f16 WMMA GEMM: C[M,N] = A[M,K] @ Bt[N,K]^T (+bias) -----------
// Block = 4 waves; each wave computes a 32x64 strip (2 M-tiles x 4 N-tiles,
// 8 WMMAs per k-step sharing each B tile). grid.z batches (A/C strides).
template <bool F16OUT, bool BIAS>
__global__ __launch_bounds__(128)
void gemm_kernel(const _Float16* __restrict__ A, const _Float16* __restrict__ Bt,
                 void* __restrict__ Cv, const float* __restrict__ bias,
                 int K, int N16tot, int ldc, long Astride, long Cstride) {
    int lane = threadIdx.x & 31, w = threadIdx.x >> 5;
    int row = lane & 15, hi = lane >> 4;
    int mbase = blockIdx.x * 32;
    int n16base = blockIdx.y * 16 + w * 4;
    A += (long)blockIdx.z * Astride;
    long coff = (long)blockIdx.z * Cstride;

    v8f acc[2][4] = {};
    for (int kc = 0; kc < K; kc += 32) {
        v16h a0 = ld_tile(A + (long)(mbase + row) * K + kc, hi);
        v16h a1 = ld_tile(A + (long)(mbase + 16 + row) * K + kc, hi);
#pragma unroll
        for (int j = 0; j < 4; ++j) {
            int n16 = n16base + j;
            if (n16 >= N16tot) continue;
            v16h b = ld_tile(Bt + (long)(n16 * 16 + row) * K + kc, hi);
            acc[0][j] = wmma_f16(a0, b, acc[0][j]);
            acc[1][j] = wmma_f16(a1, b, acc[1][j]);
        }
    }
#pragma unroll
    for (int mi = 0; mi < 2; ++mi)
#pragma unroll
        for (int j = 0; j < 4; ++j) {
            int n16 = n16base + j;
            if (n16 >= N16tot) continue;
            int n = n16 * 16 + row;
#pragma unroll
            for (int jj = 0; jj < 8; ++jj) {
                int m = mbase + mi * 16 + hi * 8 + jj;
                float v = acc[mi][j][jj];
                if (BIAS) v += bias[n];
                if (F16OUT)
                    ((_Float16*)Cv)[coff + (long)m * ldc + n] = (_Float16)v;
                else
                    ((float*)Cv)[coff + (long)m * ldc + n] = v;
            }
        }
}

// ---- scores: S[bh,q,t] = (Q·K + QP[q, clip(q-t)+512]) * hd^-0.5 ----------
// Block = 4 waves, one 64-query group; each wave holds 4 query A-tile pairs
// in registers and sweeps t-tiles (stride 4): 8 WMMAs per K-tile load pair.
__global__ __launch_bounds__(128)
void score_kernel(const _Float16* __restrict__ Qr, const _Float16* __restrict__ Kr,
                  const _Float16* __restrict__ QP, _Float16* __restrict__ Sb) {
    int lane = threadIdx.x & 31, w = threadIdx.x >> 5;
    int row = lane & 15, hi = lane >> 4;
    int bh = blockIdx.z;
    int qgroup = blockIdx.x * 64;

    v16h a[4][2];
#pragma unroll
    for (int qi = 0; qi < 4; ++qi) {
        const _Float16* qb = Qr + ((long)bh * SEQ + qgroup + qi * 16) * HDIM;
        a[qi][0] = ld_tile(qb + row * HDIM, hi);
        a[qi][1] = ld_tile(qb + row * HDIM + 32, hi);
    }

    for (int tt = w; tt < SEQ / 16; tt += 4) {
        const _Float16* kb = Kr + ((long)bh * SEQ + tt * 16) * HDIM;
        v16h b0 = ld_tile(kb + row * HDIM, hi);
        v16h b1 = ld_tile(kb + row * HDIM + 32, hi);
        int t = tt * 16 + row;
#pragma unroll
        for (int qi = 0; qi < 4; ++qi) {
            v8f acc = {};
            acc = wmma_f16(a[qi][0], b0, acc);
            acc = wmma_f16(a[qi][1], b1, acc);
#pragma unroll
            for (int jj = 0; jj < 8; ++jj) {
                int q = qgroup + qi * 16 + hi * 8 + jj;
                int dd = q - t;
                dd = dd > 512 ? 512 : (dd < -512 ? -512 : dd);
                float pos = (float)QP[((long)bh * SEQ + q) * RPAD + dd + 512];
                Sb[((long)bh * SEQ + q) * SEQ + t] =
                    (_Float16)((acc[jj] + pos) * 0.125f);
            }
        }
    }
}

// ---- row softmax over 1024 f16 values, in place ---------------------------
__global__ __launch_bounds__(128)
void softmax_kernel(_Float16* __restrict__ Sb) {
    long rowoff = (long)blockIdx.x * SEQ;
    int tid = threadIdx.x;
    v8h x = *(const v8h*)(Sb + rowoff + tid * 8);
    float v[8], m = -1e30f;
#pragma unroll
    for (int i = 0; i < 8; ++i) { v[i] = (float)x[i]; m = fmaxf(m, v[i]); }
    __shared__ float red[128];
    red[tid] = m; __syncthreads();
    for (int s = 64; s > 0; s >>= 1) {
        if (tid < s) red[tid] = fmaxf(red[tid], red[tid + s]);
        __syncthreads();
    }
    m = red[0]; __syncthreads();
    float sum = 0.f;
#pragma unroll
    for (int i = 0; i < 8; ++i) { v[i] = __expf(v[i] - m); sum += v[i]; }
    red[tid] = sum; __syncthreads();
    for (int s = 64; s > 0; s >>= 1) {
        if (tid < s) red[tid] += red[tid + s];
        __syncthreads();
    }
    float inv = 1.f / red[0];
    v8h o;
#pragma unroll
    for (int i = 0; i < 8; ++i) o[i] = (_Float16)(v[i] * inv);
    *(v8h*)(Sb + rowoff + tid * 8) = o;
}

// ---- ctx[b,q,h*64+d] = sum_t P[bh,q,t] * V[bh,t,d]  (Vt is [bh,d,t]) ------
// Each wave: 2 query tiles sharing the Vt B-tiles -> 8 WMMAs per k-step.
__global__ __launch_bounds__(128)
void pv_kernel(const _Float16* __restrict__ Sb, const _Float16* __restrict__ Vt,
               _Float16* __restrict__ ctx) {
    int lane = threadIdx.x & 31, w = threadIdx.x >> 5;
    int row = lane & 15, hi = lane >> 4;
    int bh = blockIdx.z;
    int qt0 = (blockIdx.x * 4 + w) * 2;          // 2 query tiles per wave
    int b = bh >> 3, h = bh & 7;

    const _Float16* pb0 = Sb + ((long)bh * SEQ + qt0 * 16) * SEQ;
    const _Float16* pb1 = pb0 + 16 * SEQ;
    const _Float16* vb = Vt + (long)bh * HDIM * SEQ;
    v8f acc[2][4] = {};
    for (int kc = 0; kc < SEQ; kc += 32) {
        v16h a0 = ld_tile(pb0 + (long)row * SEQ + kc, hi);
        v16h a1 = ld_tile(pb1 + (long)row * SEQ + kc, hi);
#pragma unroll
        for (int j = 0; j < 4; ++j) {
            v16h bm = ld_tile(vb + (long)(j * 16 + row) * SEQ + kc, hi);
            acc[0][j] = wmma_f16(a0, bm, acc[0][j]);
            acc[1][j] = wmma_f16(a1, bm, acc[1][j]);
        }
    }
#pragma unroll
    for (int mi = 0; mi < 2; ++mi)
#pragma unroll
        for (int j = 0; j < 4; ++j) {
            int dcol = j * 16 + row;
#pragma unroll
            for (int jj = 0; jj < 8; ++jj) {
                int q = (qt0 + mi) * 16 + hi * 8 + jj;
                ctx[((long)(b * SEQ + q)) * DMODEL + h * HDIM + dcol] =
                    (_Float16)acc[mi][j][jj];
            }
        }
}

// ---------------------------------------------------------------------------

extern "C" void kernel_launch(void* const* d_in, const int* in_sizes, int n_in,
                              void* d_out, int out_size, void* d_ws, size_t ws_size,
                              hipStream_t stream) {
    const float* x    = (const float*)d_in[0];
    // d_in[1] = attn_mask (unused, faithful to reference)
    const float* Wq   = (const float*)d_in[2];
    const float* Wkv  = (const float*)d_in[3];
    const float* Wout = (const float*)d_in[4];
    const float* bout = (const float*)d_in[5];
    const float* pemb = (const float*)d_in[6];
    float* out = (float*)d_out;

    const int TOK = BATCH * SEQ;          // 4096
    char* ws = (char*)d_ws;
    size_t off = 0;
    auto alloc = [&](size_t halves) {
        off = (off + 255) & ~(size_t)255;
        void* p = ws + off;
        off += halves * sizeof(_Float16);
        return (_Float16*)p;
    };
    _Float16* xh    = alloc((size_t)TOK * DMODEL);
    _Float16* WqT   = alloc((size_t)DMODEL * DMODEL);
    _Float16* WkvT  = alloc((size_t)2 * DMODEL * DMODEL);
    _Float16* WoutT = alloc((size_t)DMODEL * DMODEL);
    _Float16* pembh = alloc((size_t)RPAD * HDIM);
    _Float16* Qh    = alloc((size_t)TOK * DMODEL);
    _Float16* KVh   = alloc((size_t)TOK * 2 * DMODEL);
    _Float16* Qr    = alloc((size_t)BATCH * HEADS * SEQ * HDIM);
    _Float16* Kr    = alloc((size_t)BATCH * HEADS * SEQ * HDIM);
    _Float16* Vt    = alloc((size_t)BATCH * HEADS * SEQ * HDIM);
    _Float16* QP    = alloc((size_t)BATCH * HEADS * SEQ * RPAD);
    _Float16* Sb    = alloc((size_t)BATCH * HEADS * SEQ * SEQ);
    _Float16* ctx   = alloc((size_t)TOK * DMODEL);

    // --- prep: casts / transposes ---
    cast_f32_to_f16<<<(TOK * DMODEL + 255) / 256, 256, 0, stream>>>(x, xh, TOK * DMODEL);
    transpose_to_f16<<<(DMODEL * DMODEL + 255) / 256, 256, 0, stream>>>(Wq, WqT, DMODEL, DMODEL);
    transpose_to_f16<<<(DMODEL * 2 * DMODEL + 255) / 256, 256, 0, stream>>>(Wkv, WkvT, DMODEL, 2 * DMODEL);
    transpose_to_f16<<<(DMODEL * DMODEL + 255) / 256, 256, 0, stream>>>(Wout, WoutT, DMODEL, DMODEL);
    pemb_prep<<<(RPAD * HDIM + 255) / 256, 256, 0, stream>>>(pemb, pembh);

    // --- Q = x@Wq, KV = x@Wkv ---
    gemm_kernel<true, false><<<dim3(TOK / 32, DMODEL / 256, 1), 128, 0, stream>>>(
        xh, WqT, Qh, nullptr, DMODEL, DMODEL / 16, DMODEL, 0, 0);
    gemm_kernel<true, false><<<dim3(TOK / 32, 2 * DMODEL / 256, 1), 128, 0, stream>>>(
        xh, WkvT, KVh, nullptr, DMODEL, 2 * DMODEL / 16, 2 * DMODEL, 0, 0);

    rearrange_qkv<<<(BATCH * HEADS * SEQ * HDIM + 255) / 256, 256, 0, stream>>>(
        Qh, KVh, Qr, Kr, Vt);

    // --- QP = Q @ pemb^T, batched over 32 (b,h), N = 1025 padded to 1040 ---
    gemm_kernel<true, false><<<dim3(SEQ / 32, (RPAD / 16 + 15) / 16, BATCH * HEADS), 128, 0, stream>>>(
        Qr, pembh, QP, nullptr, HDIM, RPAD / 16 /* = 65 valid tiles */,
        RPAD, (long)SEQ * HDIM, (long)SEQ * RPAD);

    // --- scores (+pos gather, *scale) -> softmax (in place) -> P@V ---
    score_kernel<<<dim3(SEQ / 64, 1, BATCH * HEADS), 128, 0, stream>>>(Qr, Kr, QP, Sb);
    softmax_kernel<<<BATCH * HEADS * SEQ, 128, 0, stream>>>(Sb);
    pv_kernel<<<dim3(SEQ / 128, 1, BATCH * HEADS), 128, 0, stream>>>(Sb, Vt, ctx);

    // --- out = ctx @ Wout + bout (f32) ---
    gemm_kernel<false, true><<<dim3(TOK / 32, DMODEL / 256, 1), 128, 0, stream>>>(
        ctx, WoutT, out, bout, DMODEL, DMODEL / 16, DMODEL, 0, 0);
}
